// LidarNet_46746424049936
// MI455X (gfx1250) — compile-verified, matrix operand order checked by reading, not run on hardware
//
#include <hip/hip_runtime.h>
#include <hip/hip_bf16.h>
#include <math.h>

typedef __attribute__((ext_vector_type(16))) __bf16 v16bf;
typedef __attribute__((ext_vector_type(8)))  __bf16 v8bf;
typedef __attribute__((ext_vector_type(8)))  float  v8f;
typedef __attribute__((ext_vector_type(4)))  unsigned int v4u;
typedef __attribute__((ext_vector_type(4)))  int v4i;
typedef __attribute__((ext_vector_type(8)))  int v8i;

#define GXD 128
#define GYD 128
#define GZD 64
#define YP  130                   // y-padded stride (halo y=-1 and y=128)
#define BD  2
#define C1D 32
#define C2D 64
#define NVOX (BD*GZD*GXD*GYD)     // 2097152 logical voxels
#define NVOXP (BD*GZD*GXD*YP)     // 2129920 padded voxel cells
#define COUNT_F 2097152.0f
#define BN_EPS 1e-5f
#define W2P_ELEMS (27 * 4 * 32 * 16)   // 55296 bf16 = 110592 B

__device__ __forceinline__ float gelu_exact(float x) {
    return 0.5f * x * (1.0f + erff(x * 0.70710678118654752f));
}

// ---------------------------------------------------------------- zero
__global__ void k_zero(float* __restrict__ p, int n) {
    int i = blockIdx.x * blockDim.x + threadIdx.x;
    if (i < n) p[i] = 0.0f;
}

// ---------------------------------------------------------------- voxelize
// padded grid layout: (B, Z=64, X=128, Yp=130), logical y at index y+1
__global__ void k_voxelize(const float* __restrict__ pc, float* __restrict__ vg, int N) {
    int i = blockIdx.x * blockDim.x + threadIdx.x;
    if (i >= BD * N) return;
    int b = i / N, n = i % N;
    float x = pc[((size_t)b * 3 + 0) * N + n];
    float y = pc[((size_t)b * 3 + 1) * N + n];
    float z = pc[((size_t)b * 3 + 2) * N + n];
    if (x != x) x = 0.0f;               // nan_to_num
    if (y != y) y = 0.0f;
    if (z != z) z = 0.0f;
    float fx = floorf((x + 6.4f) / 0.1f);
    float fy = floorf((y + 6.4f) / 0.1f);
    float fz = floorf((z + 3.2f) / 0.1f);
    if (fx < 0.f || fx >= 128.f || fy < 0.f || fy >= 128.f || fz < 0.f || fz >= 64.f) return;
    int ix = (int)fx, iy = (int)fy, iz = (int)fz;
    vg[(((size_t)b * GZD + iz) * GXD + ix) * YP + iy + 1] = 1.0f;
}

// ---------------------------------------------------------------- pack W2 into per-lane B-fragment layout
// w2: (64, 32, 3,3,3) f32 -> w2p[tap][ntile][lane][16] bf16
// B (32x16, 16-bit) layout: lane<16 holds N=lane, K {0..7, 16..23};
//                           lane>=16 holds N=lane-16, K {8..15, 24..31}
__global__ void k_pack_w2(const float* __restrict__ w2, __bf16* __restrict__ w2p) {
    int tid = blockIdx.x * blockDim.x + threadIdx.x;
    if (tid >= W2P_ELEMS) return;
    int j    = tid & 15;
    int lane = (tid >> 4) & 31;
    int nt   = (tid >> 9) & 3;
    int t    = tid >> 11;
    int cout = nt * 16 + (lane & 15);
    int kbase = (lane < 16) ? 0 : 8;
    int cin  = kbase + (j < 8 ? j : j + 8);
    w2p[tid] = (__bf16)w2[((size_t)cout * C1D + cin) * 27 + t];
}

// ---------------------------------------------------------------- conv1 (1->32), two passes
template<bool WRITE_H1>
__global__ void __launch_bounds__(256) k_conv1(
    const float* __restrict__ vg,
    const float* __restrict__ w1, const float* __restrict__ b1,
    const float* __restrict__ scale1, const float* __restrict__ shift1,
    float* __restrict__ statSum, float* __restrict__ statSq,
    __bf16* __restrict__ h1)
{
    __shared__ float sw[C1D * 27];
    __shared__ float sb[C1D];
    __shared__ float ssc[C1D];
    __shared__ float ssh[C1D];
    __shared__ float sSum[C1D];
    __shared__ float sSq[C1D];

    for (int i = threadIdx.x; i < C1D * 27; i += blockDim.x) sw[i] = w1[i];
    if (threadIdx.x < C1D) {
        sb[threadIdx.x] = b1[threadIdx.x];
        if (WRITE_H1) {
            ssc[threadIdx.x] = scale1[threadIdx.x];
            ssh[threadIdx.x] = shift1[threadIdx.x];
        } else {
            sSum[threadIdx.x] = 0.0f;
            sSq[threadIdx.x]  = 0.0f;
        }
    }
    __syncthreads();

    int tid = blockIdx.x * blockDim.x + threadIdx.x;   // exactly NVOX threads
    int y = tid & 127;
    int x = (tid >> 7) & 127;
    int z = (tid >> 14) & 63;
    int b = tid >> 20;

    float acc[C1D];
#pragma unroll
    for (int c = 0; c < C1D; ++c) acc[c] = sb[c];

#pragma unroll
    for (int t = 0; t < 27; ++t) {
        int dz = t / 9 - 1, dx = (t / 3) % 3 - 1, dy = t % 3 - 1;
        int zz = z + dz, xx = x + dx;
        float v = 0.0f;
        if ((unsigned)zz < (unsigned)GZD && (unsigned)xx < (unsigned)GXD)
            v = vg[(((size_t)b * GZD + zz) * GXD + xx) * YP + (y + dy) + 1];
#pragma unroll
        for (int c = 0; c < C1D; ++c) acc[c] += sw[c * 27 + t] * v;
    }

    if (WRITE_H1) {
        size_t vox = (((size_t)b * GZD + z) * GXD + x) * YP + y + 1;
#pragma unroll
        for (int c = 0; c < C1D; ++c) {
            float g = gelu_exact(ssc[c] * acc[c] + ssh[c]);
            h1[vox * C1D + c] = (__bf16)g;
        }
    } else {
#pragma unroll
        for (int c = 0; c < C1D; ++c) {
            atomicAdd(&sSum[c], acc[c]);
            atomicAdd(&sSq[c], acc[c] * acc[c]);
        }
        __syncthreads();
        if (threadIdx.x < C1D) {
            atomicAdd(&statSum[threadIdx.x], sSum[threadIdx.x]);
            atomicAdd(&statSq[threadIdx.x],  sSq[threadIdx.x]);
        }
    }
}

// ---------------------------------------------------------------- BN finalize
__global__ void k_finalize(const float* __restrict__ sum, const float* __restrict__ sq,
                           const float* __restrict__ g, const float* __restrict__ be,
                           float* __restrict__ scale, float* __restrict__ shift,
                           int C, float count) {
    int c = threadIdx.x;
    if (c >= C) return;
    float m  = sum[c] / count;
    float v  = sq[c] / count - m * m;
    float sc = g[c] * rsqrtf(v + BN_EPS);
    scale[c] = sc;
    shift[c] = be[c] - m * sc;
}

// ---------------------------------------------------------------- conv2 (32->64) via WMMA bf16
// One block per (b,z,x); 4 waves x 32 y-positions (2 M-tiles per wave).
// Weights staged to LDS by the Tensor Data Mover (one descriptor per block),
// then per tap: 2 A fragments (unpredicated b128 pairs from y-padded h1),
// 4 B fragments from LDS, 8 independent v_wmma_f32_16x16x32_bf16.
// 27*8 = 216 WMMA per wave.
__global__ void __launch_bounds__(128) k_conv2_wmma(
    const __bf16* __restrict__ h1, const __bf16* __restrict__ w2p,
    const float* __restrict__ b2,
    float* __restrict__ pre, float* __restrict__ statSum, float* __restrict__ statSq)
{
    extern __shared__ __bf16 smW[];        // 55296 bf16 = 110592 B
    __shared__ float sSum[C2D];
    __shared__ float sSq[C2D];
    __shared__ float sB[C2D];

    const int tidb = threadIdx.x;
    const int lane = tidb & 31;
    const int wave = tidb >> 5;

#if defined(__AMDGCN__) && __has_builtin(__builtin_amdgcn_tensor_load_to_lds) && \
    __has_builtin(__builtin_amdgcn_s_wait_tensorcnt)
    // --- TDM path: one async 110592-byte global->LDS tensor copy per block ---
    if (wave == 0) {
        const unsigned           lds = (unsigned)(size_t)smW;       // LDS byte offset
        const unsigned long long ga  = (unsigned long long)(size_t)w2p;
        v4u g0;
        g0[0] = 1u;                                   // count=1, user descriptor
        g0[1] = lds;                                  // lds_addr
        g0[2] = (unsigned)ga;                         // global_addr[31:0]
        g0[3] = (unsigned)((ga >> 32) & 0x01FFFFFFu)  // global_addr[56:32]
                | (2u << 30);                         // type=2 ("image")
        v8i g1;
        g1[0] = (int)(1u << 16);                      // data_size=1 (2B), mask=0
        g1[1] = (int)((unsigned)W2P_ELEMS << 16);     // tensor_dim0[15:0]
        g1[2] = (int)((unsigned)W2P_ELEMS >> 16);     // tensor_dim0[31:16]
        g1[3] = (int)((unsigned)W2P_ELEMS << 16);     // tile_dim0 = 55296
        g1[4] = 0;                                    // tile_dim1/2 unused (1 row)
        g1[5] = (int)W2P_ELEMS;                       // tensor_dim0_stride lo
        g1[6] = 0;
        g1[7] = 0;
        v4i gz = {0, 0, 0, 0};
#if __clang_major__ >= 23
        v8i gz8 = {0, 0, 0, 0, 0, 0, 0, 0};
        __builtin_amdgcn_tensor_load_to_lds(g0, g1, gz, gz, gz8, 0);
#else
        __builtin_amdgcn_tensor_load_to_lds(g0, g1, gz, gz, 0);
#endif
        __builtin_amdgcn_s_wait_tensorcnt((short)0);
    }
#else
    // --- fallback: vectorized copy loop ---
    {
        const uint4* src = (const uint4*)w2p;
        uint4*       dst = (uint4*)smW;
        for (int i = tidb; i < W2P_ELEMS / 8; i += 128) dst[i] = src[i];
    }
#endif
    if (tidb < C2D) { sSum[tidb] = 0.0f; sSq[tidb] = 0.0f; sB[tidb] = b2[tidb]; }
    __syncthreads();

    const int ybase = wave * 32;           // 2 M-tiles per wave

    const int bid = blockIdx.x;            // 2*64*128 blocks
    const int x = bid & 127;
    const int z = (bid >> 7) & 63;
    const int b = bid >> 13;

    v8f acc[2][4];
#pragma unroll
    for (int mt = 0; mt < 2; ++mt)
#pragma unroll
        for (int nt = 0; nt < 4; ++nt)
#pragma unroll
            for (int i = 0; i < 8; ++i) acc[mt][nt][i] = 0.0f;

    const int m     = lane & 15;           // A-matrix row for this lane
    const int cbase = (lane < 16) ? 0 : 8; // A 16-bit layout K split

#pragma unroll
    for (int t = 0; t < 27; ++t) {
        const int dz = t / 9 - 1, dx = (t / 3) % 3 - 1, dy = t % 3 - 1;
        const int zz = z + dz, xx = x + dx;
        if ((unsigned)zz >= (unsigned)GZD || (unsigned)xx >= (unsigned)GXD)
            continue;                      // uniform skip: EXEC stays all-ones

        // unconditional A loads from y-padded h1 (halo rows are zero)
        v16bf a[2];
#pragma unroll
        for (int mt = 0; mt < 2; ++mt) {
            const int yy = ybase + mt * 16 + m + dy;   // in [-1, 128] -> padded
            const __bf16* arow =
                h1 + ((((size_t)b * GZD + zz) * GXD + xx) * YP + yy + 1) * C1D;
            v8bf* ah = reinterpret_cast<v8bf*>(&a[mt]);
            ah[0] = *(const v8bf*)(arow + cbase);
            ah[1] = *(const v8bf*)(arow + cbase + 16);
        }

#pragma unroll
        for (int nt = 0; nt < 4; ++nt) {
            const v16bf bf = *(const v16bf*)(smW + (((t * 4 + nt) * 32) + lane) * 16);
            acc[0][nt] = __builtin_amdgcn_wmma_f32_16x16x32_bf16(
                false, a[0], false, bf, (short)0, acc[0][nt], false, false);
            acc[1][nt] = __builtin_amdgcn_wmma_f32_16x16x32_bf16(
                false, a[1], false, bf, (short)0, acc[1][nt], false, false);
        }
    }

    // epilogue: +bias, store pre-BN f32 (layout b,z,x,c,y), reduce stats
    const int crow = lane & 15;
    const int mofs = (lane < 16) ? 0 : 8;  // C/D layout: vgpr r -> M=r / M=8+r
#pragma unroll
    for (int mt = 0; mt < 2; ++mt) {
#pragma unroll
        for (int nt = 0; nt < 4; ++nt) {
            const int   ch   = nt * 16 + crow;
            const float bias = sB[ch];
            float s = 0.0f, sq = 0.0f;
            float* prow = pre + ((((size_t)b * GZD + z) * GXD + x) * C2D + ch) * GYD
                              + ybase + mt * 16 + mofs;
#pragma unroll
            for (int r = 0; r < 8; ++r) {
                float v = acc[mt][nt][r] + bias;
                prow[r] = v;
                s += v; sq += v * v;
            }
            atomicAdd(&sSum[ch], s);
            atomicAdd(&sSq[ch],  sq);
        }
    }
    __syncthreads();
    if (tidb < C2D) {
        atomicAdd(&statSum[tidb], sSum[tidb]);
        atomicAdd(&statSq[tidb],  sSq[tidb]);
    }
}

// ---------------------------------------------------------------- BN2 + GELU + mean over z
__global__ void k_final(const float* __restrict__ pre,
                        const float* __restrict__ scale2, const float* __restrict__ shift2,
                        float* __restrict__ out)
{
    int tid = blockIdx.x * blockDim.x + threadIdx.x;  // exactly NVOX threads
    int y = tid & 127;
    int x = (tid >> 7) & 127;
    int c = (tid >> 14) & 63;
    int b = tid >> 20;
    float sc = scale2[c], sh = shift2[c];
    float acc = 0.0f;
    const size_t zstride = (size_t)GXD * C2D * GYD;
    const float* p = pre + ((((size_t)b * GZD + 0) * GXD + x) * C2D + c) * GYD + y;
    for (int z = 0; z < GZD; ++z) {
        __builtin_prefetch(p + zstride, 0, 0);
        acc += gelu_exact(sc * (*p) + sh);
        p += zstride;
    }
    // out layout (B, 64, 128, 128) == tid order
    out[tid] = acc * (1.0f / 64.0f);
}

// ---------------------------------------------------------------- launch
extern "C" void kernel_launch(void* const* d_in, const int* in_sizes, int n_in,
                              void* d_out, int out_size, void* d_ws, size_t ws_size,
                              hipStream_t stream) {
    const float* pc  = (const float*)d_in[0];
    const float* w1  = (const float*)d_in[1];
    const float* b1  = (const float*)d_in[2];
    const float* g1  = (const float*)d_in[3];
    const float* be1 = (const float*)d_in[4];
    const float* w2  = (const float*)d_in[5];
    const float* b2  = (const float*)d_in[6];
    const float* g2  = (const float*)d_in[7];
    const float* be2 = (const float*)d_in[8];
    const int N = in_sizes[0] / (BD * 3);

    char* ws = (char*)d_ws;
    float*  vg  = (float*) (ws + 0);            // 8.5 MB  y-padded voxel grid
    __bf16* h1  = (__bf16*)(ws + 8519680);      // 130 MB  y-padded bf16 activations
    float*  pre = (float*) (ws + 144834560);    // 512 MB  conv2 pre-BN f32 (b,z,x,c,y)
    __bf16* w2p = (__bf16*)(ws + 681705472);    // 108 KB  packed bf16 weights
    float*  st  = (float*) (ws + 681816064);    // stats
    float* s1sum = st;        float* s1sq = st + 32;
    float* sc1   = st + 64;   float* sh1  = st + 96;
    float* s2sum = st + 128;  float* s2sq = st + 192;
    float* sc2   = st + 256;  float* sh2  = st + 320;

    const int h1_floats = NVOXP * C1D / 2;      // bf16 buffer zero-filled as f32

    k_zero<<<(NVOXP + 255) / 256, 256, 0, stream>>>(vg, NVOXP);
    k_zero<<<(h1_floats + 255) / 256, 256, 0, stream>>>((float*)h1, h1_floats);
    k_zero<<<2, 256, 0, stream>>>(st, 384);
    k_voxelize<<<(BD * N + 255) / 256, 256, 0, stream>>>(pc, vg, N);
    k_pack_w2<<<(W2P_ELEMS + 255) / 256, 256, 0, stream>>>(w2, w2p);

    k_conv1<false><<<NVOX / 256, 256, 0, stream>>>(vg, w1, b1, nullptr, nullptr,
                                                   s1sum, s1sq, nullptr);
    k_finalize<<<1, 64, 0, stream>>>(s1sum, s1sq, g1, be1, sc1, sh1, 32, COUNT_F);
    k_conv1<true><<<NVOX / 256, 256, 0, stream>>>(vg, w1, b1, sc1, sh1,
                                                  nullptr, nullptr, h1);

    k_conv2_wmma<<<BD * GZD * GXD, 128, 110592, stream>>>(h1, w2p, b2, pre, s2sum, s2sq);
    k_finalize<<<1, 64, 0, stream>>>(s2sum, s2sq, g2, be2, sc2, sh2, 64, COUNT_F);

    k_final<<<NVOX / 256, 256, 0, stream>>>(pre, sc2, sh2, (float*)d_out);
}
